// BiCrossAttention_16509854286639
// MI455X (gfx1250) — compile-verified, matrix operand order checked by reading, not run on hardware
//
#include <hip/hip_runtime.h>
#include <stdint.h>

// ---------- types ----------
typedef __bf16 bf16;
typedef bf16   bf16x16 __attribute__((ext_vector_type(16)));
typedef float  f32x8   __attribute__((ext_vector_type(8)));
typedef unsigned int u32x4 __attribute__((ext_vector_type(4)));
typedef int    i32x4   __attribute__((ext_vector_type(4)));
typedef int    i32x8   __attribute__((ext_vector_type(8)));

union AFrag { bf16x16 v; bf16 e[16]; };
union CAcc  { f32x8 v; float e[8]; };

#define WMMA_BF16(A, B, C) \
  __builtin_amdgcn_wmma_f32_16x16x32_bf16(false, (A), false, (B), (short)0, (C), false, false)

// K-index map for 16-bit A/B fragments (16x32), per ISA 7.12.2:
// lanes 0-15: VGPR0..3 hold K=0..7 pairs, VGPR4..7 hold K=16..23; lanes 16-31: +8.
// For fixed hi this is two contiguous 8-element (16B) runs -> ds_load_b128 pairs.
__device__ __forceinline__ int kmap(int e, int hi) {
  int v = e >> 1;
  int base = (v < 4) ? (2 * v) : (16 + 2 * (v - 4));
  return base + (e & 1) + 8 * hi;
}

// Cross-lane row reductions within 16-lane halves (wave32; masks < 16 stay in-half).
__device__ __forceinline__ float rowmax16(float v) {
#pragma unroll
  for (int m = 8; m >= 1; m >>= 1) v = fmaxf(v, __shfl_xor(v, m, 32));
  return v;
}
__device__ __forceinline__ float rowsum16(float v) {
#pragma unroll
  for (int m = 8; m >= 1; m >>= 1) v += __shfl_xor(v, m, 32);
  return v;
}

// ---------- Tensor Data Mover: 2-D bf16 tile -> LDS ----------
// D# layout per CDNA5 ISA. 6-arg builtin (clang-23).
// Loads tile_rows x tile_cols (elements) from a row-major tensor whose rows
// are row_stride elements apart; tensor_cols/tensor_rows for OOB clamping.
__device__ __forceinline__ void tdm_load_2d(const bf16* gptr, unsigned lds_off,
                                            int tile_cols, int tile_rows,
                                            int row_stride, int tensor_cols,
                                            int tensor_rows) {
  unsigned long long ga = (unsigned long long)(size_t)gptr;
  u32x4 g0;
  g0[0] = 1u;                                     // count=1, user mode, no gather
  g0[1] = lds_off;                                // lds_addr
  g0[2] = (unsigned)(ga & 0xFFFFFFFFull);         // global_addr[31:0]
  g0[3] = (unsigned)((ga >> 32) & 0x1FFFFFFull)   // global_addr[56:32]
        | (2u << 30);                             // type=2 ("image")
  i32x8 g1;
  g1[0] = (1 << 16);                              // workgroup_mask=0, data_size=1 (2B)
  g1[1] = (int)((tensor_cols & 0xFFFF) << 16);    // tensor_dim0[15:0] @bit48
  g1[2] = (int)(((tensor_cols >> 16) & 0xFFFF)    // tensor_dim0[31:16]
        | ((tensor_rows & 0xFFFF) << 16));        // tensor_dim1[15:0]
  g1[3] = (int)(((tensor_rows >> 16) & 0xFFFF)    // tensor_dim1[31:16]
        | ((tile_cols & 0xFFFF) << 16));          // tile_dim0
  g1[4] = tile_rows & 0xFFFF;                     // tile_dim1 (tile_dim2=0)
  g1[5] = row_stride;                             // tensor_dim0_stride[31:0]
  g1[6] = 0;
  g1[7] = 0;
  i32x4 z4 = {0, 0, 0, 0};
  i32x8 z8 = {0, 0, 0, 0, 0, 0, 0, 0};
  __builtin_amdgcn_tensor_load_to_lds(g0, g1, z4, z4, z8, 0);
}

// =====================================================================
// Projection GEMM: Y[M=16384, 512] = X[M,512] * W[512,512]
// block: 256 thr (8 waves), tile 128x64, waves 4(M)x2(N), wave tile 32x32.
// MODE 0: f32 row-major out; MODE 1: bf16 row-major out;
// MODE 2: bf16 transposed out, per-batch [b][d][l] (for V projections).
// =====================================================================
template <int MODE>
__global__ __launch_bounds__(256) void proj_gemm_kernel(
    const float* __restrict__ X, const float* __restrict__ W,
    bf16* __restrict__ Yb, float* __restrict__ Yf) {
  constexpr int K = 512, N = 512, L = 2048;
  __shared__ bf16 sX[128][32];
  __shared__ bf16 sW[32][64];

  const int tid = threadIdx.x;
  const int lane = tid & 31;
  const int l15 = lane & 15, hi = lane >> 4;
  const int wv = tid >> 5;        // 0..7
  const int wm = wv >> 1;         // 0..3
  const int wn = wv & 1;          // 0..1

  const int mbase = (blockIdx.x >> 3) * 128;
  const int nbase = (blockIdx.x & 7) * 64;

  CAcc acc[2][2];
#pragma unroll
  for (int mi = 0; mi < 2; ++mi)
#pragma unroll
    for (int ni = 0; ni < 2; ++ni)
#pragma unroll
      for (int r = 0; r < 8; ++r) acc[mi][ni].e[r] = 0.0f;

  for (int kc = 0; kc < 16; ++kc) {
    const int kb = kc * 32;
#pragma unroll
    for (int i = 0; i < 4; ++i) {                // X tile 128x32 via float4
      int idx = tid + 256 * i;
      int r = idx >> 3, c = idx & 7;
      float4 xv = *(const float4*)&X[(size_t)(mbase + r) * K + kb + 4 * c];
      sX[r][4 * c + 0] = (bf16)xv.x;
      sX[r][4 * c + 1] = (bf16)xv.y;
      sX[r][4 * c + 2] = (bf16)xv.z;
      sX[r][4 * c + 3] = (bf16)xv.w;
    }
#pragma unroll
    for (int i = 0; i < 2; ++i) {                // W tile 32x64 via float4
      int idx = tid + 256 * i;
      int r = idx >> 4, c = idx & 15;
      float4 wv4 = *(const float4*)&W[(size_t)(kb + r) * N + nbase + 4 * c];
      sW[r][4 * c + 0] = (bf16)wv4.x;
      sW[r][4 * c + 1] = (bf16)wv4.y;
      sW[r][4 * c + 2] = (bf16)wv4.z;
      sW[r][4 * c + 3] = (bf16)wv4.w;
    }
    __syncthreads();

    AFrag a[2], b[2];
#pragma unroll
    for (int mi = 0; mi < 2; ++mi)
#pragma unroll
      for (int e = 0; e < 16; ++e)
        a[mi].e[e] = sX[32 * wm + 16 * mi + l15][kmap(e, hi)];
#pragma unroll
    for (int ni = 0; ni < 2; ++ni)
#pragma unroll
      for (int e = 0; e < 16; ++e)
        b[ni].e[e] = sW[kmap(e, hi)][32 * wn + 16 * ni + l15];

#pragma unroll
    for (int mi = 0; mi < 2; ++mi)
#pragma unroll
      for (int ni = 0; ni < 2; ++ni)
        acc[mi][ni].v = WMMA_BF16(a[mi].v, b[ni].v, acc[mi][ni].v);
    __syncthreads();
  }

#pragma unroll
  for (int mi = 0; mi < 2; ++mi)
#pragma unroll
    for (int ni = 0; ni < 2; ++ni)
#pragma unroll
      for (int r = 0; r < 8; ++r) {
        int row = mbase + 32 * wm + 16 * mi + r + 8 * hi;
        int col = nbase + 32 * wn + 16 * ni + l15;
        float val = acc[mi][ni].e[r];
        if constexpr (MODE == 0) {
          Yf[(size_t)row * N + col] = val;
        } else if constexpr (MODE == 1) {
          Yb[(size_t)row * N + col] = (bf16)val;
        } else {
          int bb = row >> 11, ll = row & (L - 1);   // batch, seq within batch
          Yb[((size_t)bb * N + col) * L + ll] = (bf16)val;
        }
      }
}

// =====================================================================
// Fused bi-branch flash attention, double-buffered TDM pipeline.
// block: 128 thr (4 waves) handles 16 query rows of one batch.
// wave w owns output columns [128w, 128w+128).
// out = gamma * (O1/l1 + O2/l2) + qs
// K is staged row-major [key][d]; V is staged TRANSPOSED [d][key] so all
// fragment builds are contiguous ds_load_b128 pairs.
// =====================================================================
__global__ __launch_bounds__(128) void biattn_kernel(
    const bf16* __restrict__ qs1, const bf16* __restrict__ ks1, const bf16* __restrict__ vs1t,
    const bf16* __restrict__ qs2, const bf16* __restrict__ ks2, const bf16* __restrict__ vs2t,
    const float* __restrict__ qsf, const float* __restrict__ gamma,
    float* __restrict__ out) {
  constexpr int L = 2048, D = 512, KT = 32, NT = L / KT;
  __shared__ bf16 sQ[16][512];        // 16 KB
  __shared__ bf16 sK[2][KT][512];     // 64 KB  (double buffered)
  __shared__ bf16 sVt[2][D][KT];      // 64 KB  (transposed, double buffered)
  __shared__ bf16 sP[4][16][KT];      // 4 KB   (wave-private P tiles)

  const int tid = threadIdx.x;
  const int lane = tid & 31;
  const int l15 = lane & 15, hi = lane >> 4;
  const int wv = tid >> 5;                 // 0..3
  const int dbase = wv * 128;

  const int b = blockIdx.x >> 7;           // batch
  const int qbase = (blockIdx.x & 127) * 16;
  const size_t boff = (size_t)b * L * D;   // row-major per-batch offset

  CAcc tot[8];
#pragma unroll
  for (int nt = 0; nt < 8; ++nt)
#pragma unroll
    for (int r = 0; r < 8; ++r) tot[nt].e[r] = 0.0f;

  for (int br = 0; br < 2; ++br) {
    const bf16* Qp = br ? qs2 : qs1;
    const bf16* Kp = br ? ks2 : ks1;
    const bf16* Vt = br ? vs2t : vs1t;     // [b][d][l] layout

    // ---- stage Q tile + preload tile 0 (TDM, wave 0) ----
    __syncthreads();                        // prior compute done before overwrite
    if (tid < 32) {
      tdm_load_2d(Qp + boff + (size_t)qbase * D, (unsigned)(size_t)&sQ[0][0],
                  D, 16, D, D, L);
      tdm_load_2d(Kp + boff, (unsigned)(size_t)&sK[0][0][0],
                  D, KT, D, D, L);
      tdm_load_2d(Vt + boff, (unsigned)(size_t)&sVt[0][0][0],
                  KT, D, L, L, D);
      __builtin_amdgcn_s_wait_tensorcnt(0);
    }
    __syncthreads();

    CAcc acc[8];
    float m_run[8], l_run[8];
#pragma unroll
    for (int r = 0; r < 8; ++r) { m_run[r] = -1e30f; l_run[r] = 0.0f; }
#pragma unroll
    for (int nt = 0; nt < 8; ++nt)
#pragma unroll
      for (int r = 0; r < 8; ++r) acc[nt].e[r] = 0.0f;

    for (int kt = 0; kt < NT; ++kt) {
      const int cur = kt & 1;
      // issue next tile's DMA into the other buffer (overlaps with compute)
      if (tid < 32 && (kt + 1) < NT) {
        tdm_load_2d(Kp + boff + (size_t)(kt + 1) * KT * D,
                    (unsigned)(size_t)&sK[cur ^ 1][0][0], D, KT, D, D, L);
        tdm_load_2d(Vt + boff + (size_t)(kt + 1) * KT,
                    (unsigned)(size_t)&sVt[cur ^ 1][0][0], KT, D, L, L, D);
      }

      // ---- S = Q * K^T  (16 x 32 tile, each wave redundantly) ----
      CAcc s0, s1;
#pragma unroll
      for (int r = 0; r < 8; ++r) { s0.e[r] = 0.0f; s1.e[r] = 0.0f; }
      for (int kc = 0; kc < 16; ++kc) {
        const int kb = kc * 32;
        AFrag aq, b0, b1;
#pragma unroll
        for (int e = 0; e < 16; ++e) {
          int kk = kb + kmap(e, hi);
          aq.e[e] = sQ[l15][kk];
          b0.e[e] = sK[cur][l15][kk];        // B(kk=d, n=key): keys 0..15
          b1.e[e] = sK[cur][16 + l15][kk];   // keys 16..31
        }
        s0.v = WMMA_BF16(aq.v, b0.v, s0.v);
        s1.v = WMMA_BF16(aq.v, b1.v, s1.v);
      }

      // ---- online softmax (per-row state in registers) ----
#pragma unroll
      for (int r = 0; r < 8; ++r) {
        float tmax = rowmax16(fmaxf(s0.e[r], s1.e[r]));
        float m_new = fmaxf(m_run[r], tmax);
        float corr = __expf(m_run[r] - m_new);
        float p0 = __expf(s0.e[r] - m_new);
        float p1 = __expf(s1.e[r] - m_new);
        float rs = rowsum16(p0 + p1);
        l_run[r] = l_run[r] * corr + rs;
        m_run[r] = m_new;
        s0.e[r] = p0; s1.e[r] = p1;
#pragma unroll
        for (int nt = 0; nt < 8; ++nt) acc[nt].e[r] *= corr;
      }

      // ---- stash P as bf16 in wave-private LDS (C-layout -> row major) ----
#pragma unroll
      for (int r = 0; r < 8; ++r) {
        sP[wv][r + 8 * hi][l15]      = (bf16)s0.e[r];
        sP[wv][r + 8 * hi][16 + l15] = (bf16)s1.e[r];
      }

      // ---- O += P * V  (K=32 keys, 8 N-tiles of this wave's D-slice) ----
      AFrag ap;
#pragma unroll
      for (int e = 0; e < 16; ++e) ap.e[e] = sP[wv][l15][kmap(e, hi)];
#pragma unroll
      for (int nt = 0; nt < 8; ++nt) {
        AFrag bv;
#pragma unroll
        for (int e = 0; e < 16; ++e)          // contiguous: sVt row = 64B
          bv.e[e] = sVt[cur][dbase + nt * 16 + l15][kmap(e, hi)];
        acc[nt].v = WMMA_BF16(ap.v, bv.v, acc[nt].v);
      }

      __syncthreads();                        // everyone done with buffer `cur`
      if (tid < 32) __builtin_amdgcn_s_wait_tensorcnt(0);
      __syncthreads();                        // next buffer is ready
    }

    // ---- normalize branch output, accumulate ----
#pragma unroll
    for (int nt = 0; nt < 8; ++nt)
#pragma unroll
      for (int r = 0; r < 8; ++r) tot[nt].e[r] += acc[nt].e[r] / l_run[r];
  }

  // ---- epilogue: out = gamma * o + qs ----
  const float g = gamma[0];
#pragma unroll
  for (int nt = 0; nt < 8; ++nt)
#pragma unroll
    for (int r = 0; r < 8; ++r) {
      int row = qbase + r + 8 * hi;
      int col = dbase + nt * 16 + l15;
      size_t idx = boff + (size_t)row * D + col;
      out[idx] = g * tot[nt].e[r] + qsf[idx];
    }
}

// =====================================================================
extern "C" void kernel_launch(void* const* d_in, const int* in_sizes, int n_in,
                              void* d_out, int out_size, void* d_ws, size_t ws_size,
                              hipStream_t stream) {
  (void)in_sizes; (void)n_in; (void)out_size; (void)ws_size;
  const float* q     = (const float*)d_in[0];
  const float* k1    = (const float*)d_in[1];
  const float* v1    = (const float*)d_in[2];
  const float* k2    = (const float*)d_in[3];
  const float* v2    = (const float*)d_in[4];
  const float* w_qs  = (const float*)d_in[5];
  const float* w_qs1 = (const float*)d_in[6];
  const float* w_qs2 = (const float*)d_in[7];
  const float* w_ks1 = (const float*)d_in[8];
  const float* w_ks2 = (const float*)d_in[9];
  const float* w_vs1 = (const float*)d_in[10];
  const float* w_vs2 = (const float*)d_in[11];
  const float* gamma = (const float*)d_in[12];
  float* out = (float*)d_out;

  char* ws = (char*)d_ws;
  const size_t MB = 1ull << 20;
  float* qs_f   = (float*)(ws);              // 32 MB f32
  bf16*  qs1_b  = (bf16*)(ws + 32 * MB);     // 16 MB bf16 each below
  bf16*  qs2_b  = (bf16*)(ws + 48 * MB);
  bf16*  ks1_b  = (bf16*)(ws + 64 * MB);
  bf16*  ks2_b  = (bf16*)(ws + 80 * MB);
  bf16*  vs1_bt = (bf16*)(ws + 96 * MB);     // transposed [b][d][l]
  bf16*  vs2_bt = (bf16*)(ws + 112 * MB);

  // 7 projections: M=16384 rows, grid = (16384/128) * (512/64) = 1024 blocks.
  dim3 pg(1024), pb(256);
  proj_gemm_kernel<0><<<pg, pb, 0, stream>>>(q,  w_qs,  (bf16*)nullptr, qs_f);
  proj_gemm_kernel<1><<<pg, pb, 0, stream>>>(q,  w_qs1, qs1_b,  (float*)nullptr);
  proj_gemm_kernel<1><<<pg, pb, 0, stream>>>(q,  w_qs2, qs2_b,  (float*)nullptr);
  proj_gemm_kernel<1><<<pg, pb, 0, stream>>>(k1, w_ks1, ks1_b,  (float*)nullptr);
  proj_gemm_kernel<1><<<pg, pb, 0, stream>>>(k2, w_ks2, ks2_b,  (float*)nullptr);
  proj_gemm_kernel<2><<<pg, pb, 0, stream>>>(v1, w_vs1, vs1_bt, (float*)nullptr);
  proj_gemm_kernel<2><<<pg, pb, 0, stream>>>(v2, w_vs2, vs2_bt, (float*)nullptr);

  // Fused bi-branch flash attention: 8 batches * 128 query tiles = 1024 blocks.
  biattn_kernel<<<dim3(1024), dim3(128), 0, stream>>>(
      qs1_b, ks1_b, vs1_bt, qs2_b, ks2_b, vs2_bt, qs_f, gamma, out);
}
